// ContextAwareAttentionBlock_12180527251476
// MI455X (gfx1250) — matrix operand order repaired and checked by reading, compiled, not run on hardware
//
#include <hip/hip_runtime.h>
#include <hip/hip_bf16.h>

// ---------------------------------------------------------------------------
// ContextAwareAttentionBlock, fused flash-attention implementation for gfx1250
// B=8, C=256, H=W=64 (N=4096), FD=32, HID=128
// ---------------------------------------------------------------------------

typedef __bf16 bf16_t;
typedef __attribute__((ext_vector_type(16))) __bf16 bf16x16;
typedef __attribute__((ext_vector_type(8)))  float  f32x8;

typedef unsigned int u32x4v __attribute__((ext_vector_type(4)));
typedef int          i32x4v __attribute__((ext_vector_type(4)));
typedef int          i32x8v __attribute__((ext_vector_type(8)));

constexpr int BB  = 8;
constexpr int CC  = 256;
constexpr int NN  = 4096;   // 64*64
constexpr int FD  = 32;
constexpr int HID = 128;

#if __has_builtin(__builtin_amdgcn_tensor_load_to_lds)
#define HAVE_TDM 1
#endif

union ABFrag {
    bf16x16      v;
    unsigned int u[8];
    bf16_t       h[16];
};

__device__ inline unsigned int pack2bf(float a, float b) {
    union { bf16_t h[2]; unsigned int u; } t;
    t.h[0] = (bf16_t)a;
    t.h[1] = (bf16_t)b;
    return t.u;
}

__device__ inline f32x8 zero8() {
    f32x8 z = {0.f, 0.f, 0.f, 0.f, 0.f, 0.f, 0.f, 0.f};
    return z;
}

__device__ inline void load_bfrag16(ABFrag& f, const bf16_t* p) {
    // bf16 p[0..7] and p[16..23] -> 8 dwords (B-operand table-K pattern)
    const uint4* u = (const uint4*)p;
    const uint4 u0 = u[0], u1 = u[2];
    f.u[0] = u0.x; f.u[1] = u0.y; f.u[2] = u0.z; f.u[3] = u0.w;
    f.u[4] = u1.x; f.u[5] = u1.y; f.u[6] = u1.z; f.u[7] = u1.w;
}

#define WMMA_BF16(A, B, C_) \
    __builtin_amdgcn_wmma_f32_16x16x32_bf16(false, (A), false, (B), (short)0, (C_), false, false)

// ---------------------------------------------------------------------------
// Kernel 0: one-shot f32 -> bf16 conversion of all weight matrices
// (row-major layouts preserved; subsequent kernels load fragments directly).
// grid = 256 blocks x 256 threads covers max size (wv: 65536 elements).
// ---------------------------------------------------------------------------
__global__ __launch_bounds__(256) void cvt_weights_kernel(
    const float* __restrict__ wq, const float* __restrict__ wk,
    const float* __restrict__ wv, const float* __restrict__ fcw,
    bf16_t* __restrict__ wqb, bf16_t* __restrict__ wkb,
    bf16_t* __restrict__ wvb, bf16_t* __restrict__ fcwb)
{
    const int i = blockIdx.x * 256 + threadIdx.x;
    if (i < FD * CC)  { wqb[i] = (bf16_t)wq[i]; wkb[i] = (bf16_t)wk[i]; }
    if (i < HID * CC) { fcwb[i] = (bf16_t)fcw[i]; }
    if (i < CC * CC)  { wvb[i] = (bf16_t)wv[i]; }
}

// ---------------------------------------------------------------------------
// Kernel 1: fused q/k/v projection.
// Phase A: q,kT = x^T @ w^T        (A = x^T pixel-rows, B = weights)
//          -> q [B,N,FD], kT [B,N,FD], row-major bf16
// Phase B: vT  = wv @ x  (+bv)     (A = wv rows, B = x)
//          -> vT [B,C,N] bf16  (rows contiguous along pixels -> TDM-friendly)
// grid = B * N/128 blocks, 256 threads (8 waves x 16 pixels each)
// ---------------------------------------------------------------------------
__global__ __launch_bounds__(256) void qkv_kernel(
    const float* __restrict__ x,
    const bf16_t* __restrict__ wqb, const float* __restrict__ bq,
    const bf16_t* __restrict__ wkb, const float* __restrict__ bk,
    const bf16_t* __restrict__ wvb, const float* __restrict__ bv,
    bf16_t* __restrict__ qg, bf16_t* __restrict__ ktg, bf16_t* __restrict__ vtg)
{
    const int tid   = threadIdx.x;
    const int wid   = tid >> 5;
    const int lane  = tid & 31;
    const int l16   = lane & 15;
    const int lhalf = lane >> 4;

    const int blk     = blockIdx.x;       // B * 32
    const int b       = blk >> 5;
    const int pixbase = (blk & 31) * 128 + wid * 16;

    const float* xb = x + (size_t)b * CC * NN;
    const int m = pixbase + l16;

    // ---------------- Phase A: q and kT ----------------
    {
        // A layout (16-bit 16x32): elem e: g=e/2,h=e%2 ->
        //   K = (g<4?0:16) + (g&3)*2 + h + 8*lhalf,  M = lane%16
        ABFrag afrag[8];
#pragma unroll
        for (int kc = 0; kc < 8; ++kc) {
#pragma unroll
            for (int g = 0; g < 8; ++g) {
                const int c0 = kc * 32 + ((g < 4) ? 0 : 16) + (g & 3) * 2 + lhalf * 8;
                afrag[kc].u[g] = pack2bf(xb[(size_t)c0 * NN + m], xb[(size_t)(c0 + 1) * NN + m]);
            }
        }
#pragma unroll 1
        for (int t = 0; t < 4; ++t) {
            const bf16_t* wmat = (t < 2) ? wqb : wkb;
            const float*  bias = (t < 2) ? bq : bk;
            bf16_t*       dst  = (t < 2) ? qg : ktg;
            const int     ocl  = (t & 1) * 16;

            f32x8 acc = zero8();
#pragma unroll
            for (int kc = 0; kc < 8; ++kc) {
                // B layout (16-bit 32x16): elem e -> K = (e<8 ? e : e+8) + 8*lhalf, Ncol = lane%16
                ABFrag bfrag;
                load_bfrag16(bfrag, wmat + (size_t)(ocl + l16) * CC + kc * 32 + lhalf * 8);
                acc = WMMA_BF16(afrag[kc].v, bfrag.v, acc);
            }
            const float biasv = bias[ocl + l16];
#pragma unroll
            for (int r = 0; r < 8; ++r) {
                const int row = pixbase + r + lhalf * 8;
                dst[((size_t)b * NN + row) * FD + ocl + l16] = (bf16_t)(acc[r] + biasv);
            }
        }
    }

    // ---------------- Phase B: vT = wv @ x + bv ----------------
    {
        // x as B operand: elem e -> K(channel) = (e<8 ? e : e+8) + 8*lhalf, Ncol(pixel) = lane%16
        ABFrag xfrag[8];
#pragma unroll
        for (int kc = 0; kc < 8; ++kc) {
#pragma unroll
            for (int g = 0; g < 8; ++g) {
                const int c0 = kc * 32 + ((g < 4) ? 0 : 16) + (g & 3) * 2 + lhalf * 8;
                xfrag[kc].u[g] = pack2bf(xb[(size_t)c0 * NN + m], xb[(size_t)(c0 + 1) * NN + m]);
            }
        }
#pragma unroll 1
        for (int mt = 0; mt < 16; ++mt) {       // 16 tiles of 16 output channels
            f32x8 acc = zero8();
#pragma unroll
            for (int kc = 0; kc < 8; ++kc) {
                ABFrag wa;   // wv rows as A operand (scrambled-K dword pattern)
                const unsigned int* wu =
                    (const unsigned int*)(wvb + (size_t)(mt * 16 + l16) * CC + kc * 32);
#pragma unroll
                for (int g = 0; g < 8; ++g) {
                    const int k0 = ((g < 4) ? 0 : 16) + (g & 3) * 2 + lhalf * 8;
                    wa.u[g] = wu[k0 >> 1];
                }
                acc = WMMA_BF16(wa.v, xfrag[kc].v, acc);
            }
#pragma unroll
            for (int r = 0; r < 8; ++r) {
                const int crow = mt * 16 + r + lhalf * 8;
                vtg[((size_t)b * CC + crow) * NN + pixbase + l16] = (bf16_t)(acc[r] + bv[crow]);
            }
        }
    }
}

// ---------------------------------------------------------------------------
// Kernel 2: flash attention + fused pooling-MLP score.
// grid = B * N/64 blocks, 256 threads.
// Waves: wm = wid>>1 (4 x 16 query rows), wc = wid&1 (2 x 128 C cols).
// Computes S^T (A = K rows, B = Q) so the per-lane softmax state and the
// P -> A-operand conversion are register-local (no LDS transpose).
// V tile [256 c x 64 j] is DMA'd into LDS by the Tensor Data Mover, issued by
// wave 0 and overlapped with the S^T WMMAs; completion via s_wait_tensorcnt.
// ---------------------------------------------------------------------------
__global__ __launch_bounds__(256) void flash_kernel(
    const bf16_t* __restrict__ qg, const bf16_t* __restrict__ ktg,
    const bf16_t* __restrict__ vtg, const float* __restrict__ x,
    const float* __restrict__ gamma_p,
    const bf16_t* __restrict__ fcwb, const float* __restrict__ fcb,
    const float* __restrict__ ctxw,
    float* __restrict__ score_g)
{
    __shared__ bf16_t vT[CC][64];        // 32 KB: V tile (TDM dest); reused as h_sa

    const int tid   = threadIdx.x;
    const int wid   = tid >> 5;
    const int lane  = tid & 31;
    const int l16   = lane & 15;
    const int lhalf = lane >> 4;
    const int wm    = wid >> 1;
    const int wc    = wid & 1;

    const int b     = blockIdx.x >> 6;          // 64 i-tiles per batch
    const int ibase = (blockIdx.x & 63) * 64;
    const int row16 = wm * 16;

    // ---- Q as B-operand: elem e -> f = (e<8 ? e : e+8) + 8*lhalf, i-col = lane%16 ----
    ABFrag qb;
    load_bfrag16(qb, qg + ((size_t)b * NN + ibase + row16 + l16) * FD + lhalf * 8);

    f32x8 o[8];
#pragma unroll
    for (int t = 0; t < 8; ++t) o[t] = zero8();
    float m_run = -3.0e38f, l_run = 0.f;    // per-lane: row i = lane%16

    const unsigned lds_base = (unsigned)(size_t)(void*)&vT[0][0];

#pragma unroll 1
    for (int jb = 0; jb < NN; jb += 64) {
        __syncthreads();   // previous tile's vT fully consumed

        // ---- kick off async V-tile copy: vT[c][0..63] = vtg[b, c, jb+..] ----
#ifdef HAVE_TDM
        if (wid == 0) {
            const unsigned long long gaddr =
                (unsigned long long)(size_t)(const void*)(vtg + (size_t)b * CC * NN + jb);
            u32x4v g0;
            g0[0] = 1u;                                        // count = 1 descriptor
            g0[1] = lds_base;                                  // LDS byte address
            g0[2] = (unsigned)(gaddr & 0xFFFFFFFFull);         // global addr lo
            g0[3] = (unsigned)((gaddr >> 32) & 0x01FFFFFFull) | 0x80000000u; // hi | type=2
            i32x8v g1;
            g1[0] = 0x00010000;                // data_size = 1 (2-byte elements)
            g1[1] = (int)((NN & 0xFFFF) << 16);        // tensor_dim0 lo16 (bits 63:48)
            g1[2] = (int)((NN >> 16) | (CC << 16));    // tensor_dim0 hi / tensor_dim1 lo
            g1[3] = (int)((CC >> 16) | (64 << 16));    // tensor_dim1 hi / tile_dim0 = 64
            g1[4] = CC;                                 // tile_dim1 = 256, tile_dim2 = 0
            g1[5] = NN;                                 // tensor_dim0_stride = 4096
            g1[6] = 0;
            g1[7] = 0;
            const i32x4v gz4 = {0, 0, 0, 0};
            const i32x8v gz8 = {0, 0, 0, 0, 0, 0, 0, 0};
            __builtin_amdgcn_tensor_load_to_lds(g0, g1, gz4, gz4, gz8, 0);
        }
#else
        // fallback: plain vectorized copy (rows are contiguous in vtg)
#pragma unroll
        for (int it = 0; it < 8; ++it) {
            const int idx = tid + it * 256;            // 2048 x 16B
            const int c   = idx >> 3;
            const int sg  = (idx & 7) * 8;
            *(uint4*)&vT[c][sg] = *(const uint4*)(vtg + ((size_t)b * CC + c) * NN + jb + sg);
        }
#endif
        if (jb + 64 < NN)   // prefetch next K tile (global_prefetch_b8)
            __builtin_prefetch((const void*)(ktg + ((size_t)b * NN + jb + 64 + lane * 2) * FD), 0, 0);

        // ---- S^T = K_tile @ Q : D[m=j, n=i]  (does not touch vT) ----
        f32x8 st[4];
#pragma unroll
        for (int ns = 0; ns < 4; ++ns) {
            ABFrag ka;  // kT rows as A operand, scrambled-K dword pattern
            const unsigned int* ku =
                (const unsigned int*)(ktg + ((size_t)b * NN + jb + ns * 16 + l16) * FD);
#pragma unroll
            for (int g = 0; g < 8; ++g) {
                const int k0 = ((g < 4) ? 0 : 16) + (g & 3) * 2 + lhalf * 8;
                ka.u[g] = ku[k0 >> 1];
            }
            st[ns] = WMMA_BF16(ka.v, qb.v, zero8());
        }

        // ---- online softmax: per-lane row stats (i = lane%16) ----
        // lane covers j = ns*16 + r + 8*lhalf; pair-lane (xor 16) has the rest
        float cmax = -3.0e38f;
#pragma unroll
        for (int ns = 0; ns < 4; ++ns)
#pragma unroll
            for (int r = 0; r < 8; ++r) cmax = fmaxf(cmax, st[ns][r]);
        cmax = fmaxf(cmax, __shfl_xor(cmax, 16));
        const float nm    = fmaxf(m_run, cmax);
        const float alpha = __expf(m_run - nm);
        float rs = 0.f;
#pragma unroll
        for (int ns = 0; ns < 4; ++ns)
#pragma unroll
            for (int r = 0; r < 8; ++r) {
                const float p = __expf(st[ns][r] - nm);
                st[ns][r] = p;
                rs += p;
            }
        rs += __shfl_xor(rs, 16);
        l_run = l_run * alpha + rs;
        m_run = nm;

        // broadcast alpha(i) to O-frag rows (i = r + 8*lhalf) and rescale O
        float av[8];
#pragma unroll
        for (int r = 0; r < 8; ++r) av[r] = __shfl(alpha, r + 8 * lhalf);
#pragma unroll
        for (int t = 0; t < 8; ++t)
#pragma unroll
            for (int r = 0; r < 8; ++r) o[t][r] *= av[r];

        // ---- P^T (C/D layout) -> P as A operand: pure register packing ----
        ABFrag pa[2];
#pragma unroll
        for (int kc = 0; kc < 2; ++kc)
#pragma unroll
            for (int g = 0; g < 8; ++g) {
                const f32x8& src = st[2 * kc + ((g >= 4) ? 1 : 0)];
                pa[kc].u[g] = pack2bf(src[(g & 3) * 2], src[(g & 3) * 2 + 1]);
            }

        // ---- wait for V tile, then O += P @ V ----
#ifdef HAVE_TDM
        if (wid == 0) __builtin_amdgcn_s_wait_tensorcnt(0);
#endif
        __syncthreads();

#pragma unroll
        for (int t = 0; t < 8; ++t) {
            const int cb = wc * 128 + t * 16;
#pragma unroll
            for (int kc = 0; kc < 2; ++kc) {
                ABFrag vb;
                load_bfrag16(vb, &vT[cb + l16][kc * 32 + lhalf * 8]);
                o[t] = WMMA_BF16(pa[kc].v, vb.v, o[t]);
            }
        }
    }

    // ---- epilogue: h_sa = gamma * (O/l) + x, kept in LDS (reuse vT) ----
    const float gamma = gamma_p[0];
    float lr[8];
#pragma unroll
    for (int r = 0; r < 8; ++r) lr[r] = __shfl(l_run, r + 8 * lhalf);

    __syncthreads();
    bf16_t (*hs)[CC] = (bf16_t(*)[CC]) & vT[0][0];   // hs[64][256]
#pragma unroll
    for (int t = 0; t < 8; ++t) {
        const int ccol = wc * 128 + t * 16 + l16;
#pragma unroll
        for (int r = 0; r < 8; ++r) {
            const int row = row16 + r + lhalf * 8;
            const float ov = o[t][r] / lr[r];
            const float xv = x[((size_t)b * CC + ccol) * NN + ibase + row];
            hs[row][ccol] = (bf16_t)(gamma * ov + xv);
        }
    }
    __syncthreads();

    // ---- pooling MLP: hid = tanh(h_sa @ fcw^T + fcb); score = hid . ctxw ----
    if (wid < 4) {
        const int row0 = wid * 16;
        ABFrag ha[8];
#pragma unroll
        for (int kc = 0; kc < 8; ++kc) {
            const unsigned int* pu = (const unsigned int*)&hs[row0 + l16][kc * 32];
#pragma unroll
            for (int g = 0; g < 8; ++g) {
                const int k0 = ((g < 4) ? 0 : 16) + (g & 3) * 2 + lhalf * 8;
                ha[kc].u[g] = pu[k0 >> 1];
            }
        }
        float scr[8];
#pragma unroll
        for (int r = 0; r < 8; ++r) scr[r] = 0.f;

#pragma unroll 1
        for (int t = 0; t < 8; ++t) {       // 8 tiles x 16 = 128 hidden units
            f32x8 acc = zero8();
#pragma unroll
            for (int kc = 0; kc < 8; ++kc) {
                ABFrag wb;
                load_bfrag16(wb, fcwb + (size_t)(t * 16 + l16) * CC + kc * 32 + lhalf * 8);
                acc = WMMA_BF16(ha[kc].v, wb.v, acc);
            }
            const float bias = fcb[t * 16 + l16];
            const float cw   = ctxw[t * 16 + l16];
#pragma unroll
            for (int r = 0; r < 8; ++r)
                scr[r] += cw * tanhf(acc[r] + bias);
        }
#pragma unroll
        for (int r = 0; r < 8; ++r) {
            float v = scr[r];
            v += __shfl_xor(v, 1);
            v += __shfl_xor(v, 2);
            v += __shfl_xor(v, 4);
            v += __shfl_xor(v, 8);
            if (l16 == 0) {
                const int row = row0 + r + lhalf * 8;
                score_g[(size_t)b * NN + ibase + row] = v;
            }
        }
    }
}

// ---------------------------------------------------------------------------
// Kernel 3: per-batch softmax over N scores.  grid = B, 256 threads.
// ---------------------------------------------------------------------------
__global__ __launch_bounds__(256) void softmax_kernel(
    const float* __restrict__ score, float* __restrict__ prob)
{
    __shared__ float redm[8], reds[8];
    const int b = blockIdx.x, tid = threadIdx.x;
    const float* s = score + (size_t)b * NN;

    float mx = -3.0e38f;
    for (int i = tid; i < NN; i += 256) mx = fmaxf(mx, s[i]);
#pragma unroll
    for (int m = 1; m < 32; m <<= 1) mx = fmaxf(mx, __shfl_xor(mx, m));
    if ((tid & 31) == 0) redm[tid >> 5] = mx;
    __syncthreads();
    mx = redm[0];
#pragma unroll
    for (int i = 1; i < 8; ++i) mx = fmaxf(mx, redm[i]);

    float sum = 0.f;
    for (int i = tid; i < NN; i += 256) sum += __expf(s[i] - mx);
#pragma unroll
    for (int m = 1; m < 32; m <<= 1) sum += __shfl_xor(sum, m);
    if ((tid & 31) == 0) reds[tid >> 5] = sum;
    __syncthreads();
    sum = 0.f;
#pragma unroll
    for (int i = 0; i < 8; ++i) sum += reds[i];

    const float inv = 1.0f / sum;
    for (int i = tid; i < NN; i += 256)
        prob[(size_t)b * NN + i] = __expf(s[i] - mx) * inv;
}

// ---------------------------------------------------------------------------
// Kernel 4: out[b,c] = sum_n x[b,c,n] * prob[b,n].  grid = B*C, 256 threads.
// ---------------------------------------------------------------------------
__global__ __launch_bounds__(256) void out_kernel(
    const float* __restrict__ x, const float* __restrict__ prob,
    float* __restrict__ out)
{
    __shared__ float red[8];
    const int b = blockIdx.x / CC;
    const int c = blockIdx.x % CC;
    const int tid = threadIdx.x;
    const float* xr = x + ((size_t)b * CC + c) * NN;
    const float* p  = prob + (size_t)b * NN;

    float acc = 0.f;
    for (int i = tid; i < NN; i += 256) acc = fmaf(xr[i], p[i], acc);
#pragma unroll
    for (int m = 1; m < 32; m <<= 1) acc += __shfl_xor(acc, m);
    if ((tid & 31) == 0) red[tid >> 5] = acc;
    __syncthreads();
    if (tid == 0) {
        float t = 0.f;
#pragma unroll
        for (int i = 0; i < 8; ++i) t += red[i];
        out[(size_t)b * CC + c] = t;
    }
}

// ---------------------------------------------------------------------------
// Launch
// ---------------------------------------------------------------------------
extern "C" void kernel_launch(void* const* d_in, const int* in_sizes, int n_in,
                              void* d_out, int out_size, void* d_ws, size_t ws_size,
                              hipStream_t stream)
{
    (void)in_sizes; (void)n_in; (void)out_size; (void)ws_size;

    const float* x     = (const float*)d_in[0];
    const float* wq    = (const float*)d_in[1];
    const float* bq    = (const float*)d_in[2];
    const float* wk    = (const float*)d_in[3];
    const float* bk    = (const float*)d_in[4];
    const float* wv    = (const float*)d_in[5];
    const float* bv    = (const float*)d_in[6];
    const float* gamma = (const float*)d_in[7];
    const float* fcw   = (const float*)d_in[8];
    const float* fcb   = (const float*)d_in[9];
    const float* ctxw  = (const float*)d_in[10];

    char* ws = (char*)d_ws;
    const size_t Q_SZ = (size_t)BB * NN * FD * sizeof(bf16_t);   // 2 MB
    const size_t V_SZ = (size_t)BB * NN * CC * sizeof(bf16_t);   // 16 MB
    const size_t S_SZ = (size_t)BB * NN * sizeof(float);         // 128 KB
    size_t off = 0;
    bf16_t* qg    = (bf16_t*)(ws + off); off += Q_SZ;
    bf16_t* ktg   = (bf16_t*)(ws + off); off += Q_SZ;
    bf16_t* vtg   = (bf16_t*)(ws + off); off += V_SZ;
    float*  score = (float*) (ws + off); off += S_SZ;
    float*  prob  = (float*) (ws + off); off += S_SZ;
    bf16_t* wqb   = (bf16_t*)(ws + off); off += (size_t)FD * CC * sizeof(bf16_t);
    bf16_t* wkb   = (bf16_t*)(ws + off); off += (size_t)FD * CC * sizeof(bf16_t);
    bf16_t* wvb   = (bf16_t*)(ws + off); off += (size_t)CC * CC * sizeof(bf16_t);
    bf16_t* fcwb  = (bf16_t*)(ws + off); off += (size_t)HID * CC * sizeof(bf16_t);

    cvt_weights_kernel<<<(CC * CC) / 256, 256, 0, stream>>>(wq, wk, wv, fcw,
                                                            wqb, wkb, wvb, fcwb);
    qkv_kernel<<<BB * (NN / 128), 256, 0, stream>>>(x, wqb, bq, wkb, bk, wvb, bv,
                                                    qg, ktg, vtg);
    flash_kernel<<<BB * (NN / 64), 256, 0, stream>>>(qg, ktg, vtg, x, gamma,
                                                     fcwb, fcb, ctxw, score);
    softmax_kernel<<<BB, 256, 0, stream>>>(score, prob);
    out_kernel<<<BB * CC, 256, 0, stream>>>(x, prob, (float*)d_out);
}
